// TppNet_35682588295561
// MI455X (gfx1250) — matrix operand order repaired and checked by reading, not compiled, not run on hardware
//
#include <hip/hip_runtime.h>
#include <math.h>

typedef __attribute__((ext_vector_type(16))) _Float16 v16h;
typedef __attribute__((ext_vector_type(8)))  _Float16 v8h;
typedef __attribute__((ext_vector_type(8)))  float    v8f;

#define BATCH 4
#define NPTS  1000
#define KNN   8
#define NPAIR 499500   // NPTS*(NPTS-1)/2

// ---------------------------------------------------------------------------
// A-fragment loaders (16x32 f16): lane half h supplies K run
// [k0+16h, +16)  -> v16h components 0-15.
// ---------------------------------------------------------------------------
__device__ __forceinline__ v16h load_a(const float* row, int k0, int half) {
  const float4* q0 = (const float4*)(row + k0 + 16 * half);
  const float4* q1 = (const float4*)(row + k0 + 16 * half + 8);
  float4 r0 = q0[0], r1 = q0[1], r2 = q1[0], r3 = q1[1];
  v16h a;
  a[0]  = (_Float16)r0.x; a[1]  = (_Float16)r0.y;
  a[2]  = (_Float16)r0.z; a[3]  = (_Float16)r0.w;
  a[4]  = (_Float16)r1.x; a[5]  = (_Float16)r1.y;
  a[6]  = (_Float16)r1.z; a[7]  = (_Float16)r1.w;
  a[8]  = (_Float16)r2.x; a[9]  = (_Float16)r2.y;
  a[10] = (_Float16)r2.z; a[11] = (_Float16)r2.w;
  a[12] = (_Float16)r3.x; a[13] = (_Float16)r3.y;
  a[14] = (_Float16)r3.z; a[15] = (_Float16)r3.w;
  return a;
}

__device__ __forceinline__ v16h load_a(const _Float16* row, int k0, int half) {
  v8h lo = *(const v8h*)(row + k0 + 16 * half);
  v8h hi = *(const v8h*)(row + k0 + 16 * half + 8);
  return __builtin_shufflevector(lo, hi, 0, 1, 2, 3, 4, 5, 6, 7,
                                 8, 9, 10, 11, 12, 13, 14, 15);
}

// ---------------------------------------------------------------------------
// Brute-force kNN (top-8 smallest squared distance; self included; ties keep
// lower index first — matches jax top_k(-dist, 8)).
// ---------------------------------------------------------------------------
template <int D>
__global__ void knn_kernel(const float* __restrict__ x, int ldx,
                           int* __restrict__ idx, int N) {
  const int b = blockIdx.y;
  const int i = blockIdx.x * blockDim.x + threadIdx.x;
  if (i >= N) return;
  const float* xb = x + (size_t)b * N * ldx;

  float xi[D];
#pragma unroll
  for (int d = 0; d < D; ++d) xi[d] = xb[(size_t)i * ldx + d];

  float bd[KNN];
  int   bi[KNN];
#pragma unroll
  for (int k = 0; k < KNN; ++k) { bd[k] = 3.4e38f; bi[k] = 0; }

  for (int j = 0; j < N; ++j) {
    const float* xj = xb + (size_t)j * ldx;
    float dist = 0.f;
#pragma unroll 4
    for (int d = 0; d < D; ++d) {
      float df = xi[d] - xj[d];
      dist = fmaf(df, df, dist);
    }
    if (dist < bd[KNN - 1]) {
      int p = KNN - 1;
      while (p > 0 && bd[p - 1] > dist) {
        bd[p] = bd[p - 1]; bi[p] = bi[p - 1]; --p;
      }
      bd[p] = dist; bi[p] = j;
    }
  }
  int* out = idx + ((size_t)b * N + i) * KNN;
#pragma unroll
  for (int k = 0; k < KNN; ++k) out[k] = bi[k];
}

// ---------------------------------------------------------------------------
// Gather edge features into f16, row stride ldE (zero-padded beyond 2D cols).
// E[(b*N+p)*8+k] = [ x_i | x_j - x_i | 0-pad ]
// ---------------------------------------------------------------------------
__global__ void build_edge_kernel(const float* __restrict__ x, int ldx,
                                  const int* __restrict__ idx,
                                  _Float16* __restrict__ E, int N, int D,
                                  int ldE) {
  const long r   = blockIdx.x;            // 0 .. B*N*8-1
  const int  b   = (int)(r / ((long)N * KNN));
  const long rem = r - (long)b * N * KNN;
  const int  p   = (int)(rem / KNN);
  const int  j   = idx[r];
  const float* xb = x + (size_t)b * N * ldx;
  const float* xi = xb + (size_t)p * ldx;
  const float* xj = xb + (size_t)j * ldx;
  _Float16* e = E + (size_t)r * ldE;
  for (int c = threadIdx.x; c < ldE; c += blockDim.x) {
    float v = (c < D) ? xi[c] : ((c < 2 * D) ? (xj[c - D] - xi[c - D]) : 0.f);
    e[c] = (_Float16)v;
  }
}

// ---------------------------------------------------------------------------
// One-time: pad + transpose + convert weights: W (Kd x Nd f32, row-major)
// -> Wt (Np x Kp f16), Wt[n*Kp + k] = W[k*Nd + n], zero-padded.
// ---------------------------------------------------------------------------
__global__ void convert_wT_kernel(const float* __restrict__ W, int Kd, int Nd,
                                  _Float16* __restrict__ Wt, int Kp, int Np) {
  const int t = blockIdx.x * blockDim.x + threadIdx.x;
  if (t >= Np * Kp) return;
  const int n = t / Kp, k = t % Kp;
  Wt[t] = (k < Kd && n < Nd) ? (_Float16)W[k * Nd + n] : (_Float16)0.f;
}

// Zero-pad an f32 vector/matrix (used for the layer-1 bias).
__global__ void pad_f32_kernel(const float* __restrict__ W, int Kd, int Nd,
                               float* __restrict__ Wp, int Kp, int Np) {
  const int t = blockIdx.x * blockDim.x + threadIdx.x;
  if (t >= Kp * Np) return;
  const int k = t / Np, n = t % Np;
  Wp[t] = (k < Kd && n < Nd) ? W[k * Nd + n] : 0.f;
}

// ---------------------------------------------------------------------------
// Fully aligned WMMA GEMM: one wave computes a 32x32 output tile.
// REQUIRES: M % 32 == 0, Kd % 32 == 0, Nd % 32 == 0 (pre-padded buffers).
// Wt is (Nd x ldw) f16 transposed.  Y = act(X*W + bias), optional fused
// max over groups of 8 rows (the K neighbors of one point).
// ---------------------------------------------------------------------------
template <bool RELU, bool MAXPOOL8, typename AT, typename OT>
__global__ void gemm_wmma_kernel(const AT* __restrict__ X, int ldx,
                                 const _Float16* __restrict__ Wt, int ldw,
                                 const float* __restrict__ bias,
                                 OT* __restrict__ Y, int ldy,
                                 int M, int Kd) {
  const int lane = threadIdx.x;
  const int lr   = lane & 15;
  const int half = lane >> 4;
  const int m0   = blockIdx.x * 32;
  const int n0   = blockIdx.y * 32;

  const AT* Xr0 = X + (size_t)(m0 + lr) * ldx;        // rows m0..m0+15
  const AT* Xr1 = X + (size_t)(m0 + 16 + lr) * ldx;   // rows m0+16..m0+31
  const int col0 = n0 + lr;
  const int col1 = n0 + 16 + lr;
  const _Float16* Wc0 = Wt + (size_t)col0 * ldw + 16 * half;
  const _Float16* Wc1 = Wt + (size_t)col1 * ldw + 16 * half;

  v8f acc00 = {}, acc01 = {}, acc10 = {}, acc11 = {};

  for (int k0 = 0; k0 < Kd; k0 += 32) {
    if (k0 + 32 < Kd)
      __builtin_prefetch(Wc0 + k0 + 32, 0, 1);

    v16h a0 = load_a(Xr0, k0, half);
    v16h a1 = load_a(Xr1, k0, half);
    // B fragment: comp c -> K = k0 + 16*half + c : contiguous in Wt
    v16h b0 = *(const v16h*)(Wc0 + k0);
    v16h b1 = *(const v16h*)(Wc1 + k0);

    acc00 = __builtin_amdgcn_wmma_f32_16x16x32_f16(false, a0, false, b0,
                                                   (short)0, acc00, false, false);
    acc01 = __builtin_amdgcn_wmma_f32_16x16x32_f16(false, a0, false, b1,
                                                   (short)0, acc01, false, false);
    acc10 = __builtin_amdgcn_wmma_f32_16x16x32_f16(false, a1, false, b0,
                                                   (short)0, acc10, false, false);
    acc11 = __builtin_amdgcn_wmma_f32_16x16x32_f16(false, a1, false, b1,
                                                   (short)0, acc11, false, false);
  }

  const float bv0 = bias[col0];
  const float bv1 = bias[col1];
#pragma unroll
  for (int r = 0; r < 8; ++r) {
    float v00 = acc00[r] + bv0, v01 = acc01[r] + bv1;
    float v10 = acc10[r] + bv0, v11 = acc11[r] + bv1;
    if (RELU) {
      v00 = fmaxf(v00, 0.f); v01 = fmaxf(v01, 0.f);
      v10 = fmaxf(v10, 0.f); v11 = fmaxf(v11, 0.f);
    }
    acc00[r] = v00; acc01[r] = v01; acc10[r] = v10; acc11[r] = v11;
  }

  if (MAXPOOL8) {
    float m00 = acc00[0], m01 = acc01[0], m10 = acc10[0], m11 = acc11[0];
#pragma unroll
    for (int r = 1; r < 8; ++r) {
      m00 = fmaxf(m00, acc00[r]); m01 = fmaxf(m01, acc01[r]);
      m10 = fmaxf(m10, acc10[r]); m11 = fmaxf(m11, acc11[r]);
    }
    const int pt0 = (m0 >> 3) + half;       // rows m0..m0+15  -> 2 points
    const int pt1 = (m0 >> 3) + 2 + half;   // rows m0+16..+31 -> 2 points
    Y[(size_t)pt0 * ldy + col0] = (OT)m00;
    Y[(size_t)pt0 * ldy + col1] = (OT)m01;
    Y[(size_t)pt1 * ldy + col0] = (OT)m10;
    Y[(size_t)pt1 * ldy + col1] = (OT)m11;
  } else {
#pragma unroll
    for (int r = 0; r < 8; ++r) {
      const int mA = m0 + r + 8 * half;       // C layout: M = r + 8*half
      const int mB = mA + 16;
      Y[(size_t)mA * ldy + col0] = (OT)acc00[r];
      Y[(size_t)mA * ldy + col1] = (OT)acc01[r];
      Y[(size_t)mB * ldy + col0] = (OT)acc10[r];
      Y[(size_t)mB * ldy + col1] = (OT)acc11[r];
    }
  }
}

// ---------------------------------------------------------------------------
// g[b, c] = max over N points of sf2[b, i, c]   (B*32 threads total, f16)
// ---------------------------------------------------------------------------
__global__ void colmax_kernel(const _Float16* __restrict__ sf2,
                              _Float16* __restrict__ g, int N) {
  const int t = blockIdx.x * blockDim.x + threadIdx.x;
  if (t >= BATCH * 32) return;
  const int b = t >> 5, c = t & 31;
  float m = -3.4e38f;
  for (int i = 0; i < N; ++i)
    m = fmaxf(m, (float)sf2[((size_t)b * N + i) * 32 + c]);
  g[t] = (_Float16)m;
}

// ---------------------------------------------------------------------------
// Fused pairwise edge classifier. One wave handles 16 pairs.
// ef = [sf_i | sf_j | g] (96, f16) -> relu(W1)+b1 (32) -> dot w2 + b2 -> sig.
// All loads unconditional (pair index clamped); only stores are masked.
// w1t is (32 x 96) f16 transposed.
// ---------------------------------------------------------------------------
__global__ void edge_head_kernel(const _Float16* __restrict__ sf2, // (B*N,32)
                                 const _Float16* __restrict__ g,   // (B,32)
                                 const _Float16* __restrict__ w1t, // (32,96)
                                 const float* __restrict__ b1,     // (32)
                                 const float* __restrict__ w2,     // (32)
                                 const float* __restrict__ b2,     // (1)
                                 float* __restrict__ sig_out,
                                 float* __restrict__ logit_out,
                                 int N, int P) {
  const int lane = threadIdx.x;
  const int lr   = lane & 15;
  const int half = lane >> 4;
  const int b    = blockIdx.y;
  const int m0   = blockIdx.x * 16;

  // pair index for the row this lane feeds into the A fragment (clamped)
  int p  = m0 + lr;
  int pc = p < P ? p : P - 1;

  // closed-form triu(N,1) row: off(i) = i*(2N-1-i)/2
  const double a2 = (double)(2 * N - 1);
  int i = (int)((a2 - sqrt(a2 * a2 - 8.0 * (double)pc)) * 0.5);
  if (i < 0) i = 0;
  while (i > 0 && ((long)i * (2 * N - 1 - i)) / 2 > pc) --i;
  while (((long)(i + 1) * (2 * N - 1 - (i + 1))) / 2 <= pc) ++i;
  const int j = i + 1 + (pc - (int)(((long)i * (2 * N - 1 - i)) / 2));

  const _Float16* srcs[3] = {sf2 + ((size_t)b * N + i) * 32,
                             sf2 + ((size_t)b * N + j) * 32,
                             g + (size_t)b * 32};

  v8f c0 = {}, c1 = {};
#pragma unroll
  for (int ks = 0; ks < 3; ++ks) {
    v16h a = load_a(srcs[ks], 0, half);
    // B fragment: comp c -> K = 32*ks + 16*half + c : contiguous in w1t
    v16h bf0 = *(const v16h*)(w1t + (size_t)lr * 96 + ks * 32 + 16 * half);
    v16h bf1 = *(const v16h*)(w1t + (size_t)(16 + lr) * 96 + ks * 32 + 16 * half);
    c0 = __builtin_amdgcn_wmma_f32_16x16x32_f16(false, a, false, bf0,
                                                (short)0, c0, false, false);
    c1 = __builtin_amdgcn_wmma_f32_16x16x32_f16(false, a, false, bf1,
                                                (short)0, c1, false, false);
  }

  const float bv0 = b1[lr], bv1 = b1[16 + lr];
  const float w2a = w2[lr], w2b = w2[16 + lr];
  const float b2s = b2[0];

#pragma unroll
  for (int r = 0; r < 8; ++r) {
    float h0 = fmaxf(c0[r] + bv0, 0.f);       // h[M, lr]      M = r + 8*half
    float h1 = fmaxf(c1[r] + bv1, 0.f);       // h[M, 16+lr]
    float t = h0 * w2a + h1 * w2b;
    // sum over the 16 lanes of this half (columns 0..31 of h)
    t += __shfl_xor(t, 1, 32);
    t += __shfl_xor(t, 2, 32);
    t += __shfl_xor(t, 4, 32);
    t += __shfl_xor(t, 8, 32);
    if (lr == 0) {
      const int pr = m0 + r + 8 * half;
      if (pr < P) {
        const float logit = t + b2s;
        const size_t o = (size_t)b * P + pr;
        logit_out[o] = logit;
        sig_out[o]   = 1.f / (1.f + __expf(-logit));
      }
    }
  }
}

// ---------------------------------------------------------------------------
// Orchestration
// ---------------------------------------------------------------------------
extern "C" void kernel_launch(void* const* d_in, const int* in_sizes, int n_in,
                              void* d_out, int out_size, void* d_ws, size_t ws_size,
                              hipStream_t stream) {
  const int B = BATCH, N = NPTS, P = NPAIR;
  const float* pos   = (const float*)d_in[0];
  const float* c1_w1 = (const float*)d_in[1];
  const float* c1_b1 = (const float*)d_in[2];
  const float* c1_w2 = (const float*)d_in[3];
  const float* c1_b2 = (const float*)d_in[4];
  const float* c2_w1 = (const float*)d_in[5];
  const float* c2_b1 = (const float*)d_in[6];
  const float* c2_w2 = (const float*)d_in[7];
  const float* c2_b2 = (const float*)d_in[8];
  const float* c3_w1 = (const float*)d_in[9];
  const float* c3_b1 = (const float*)d_in[10];
  const float* c3_w2 = (const float*)d_in[11];
  const float* c3_b2 = (const float*)d_in[12];
  const float* s_w1  = (const float*)d_in[13];
  const float* s_b1  = (const float*)d_in[14];
  const float* s_w2  = (const float*)d_in[15];
  const float* s_b2  = (const float*)d_in[16];
  const float* e_w1  = (const float*)d_in[17];
  const float* e_b1  = (const float*)d_in[18];
  const float* e_w2  = (const float*)d_in[19];
  const float* e_b2  = (const float*)d_in[20];

  const int M  = B * NPTS * KNN;   // 32000 edge rows
  const int BN = B * NPTS;         // 4000 point rows

  // workspace carve (256B aligned chunks)
  char* ws = (char*)d_ws;
  size_t off = 0;
  auto carve = [&](size_t bytes) {
    void* p = ws + off;
    off += (bytes + 255) & ~(size_t)255;
    return p;
  };
  int*       idx   = (int*)      carve((size_t)M * sizeof(int));
  float*     xcat  = (float*)    carve((size_t)BN * 672 * sizeof(float));
  _Float16*  gbuf  = (_Float16*) carve((size_t)B * 32 * sizeof(_Float16));
  float*     c1b1p = (float*)    carve(32 * sizeof(float));
  _Float16*  c1w1t = (_Float16*) carve(32 * 32 * sizeof(_Float16));
  _Float16*  c1w2t = (_Float16*) carve(32 * 32 * sizeof(_Float16));
  _Float16*  c2w1t = (_Float16*) carve(64 * 64 * sizeof(_Float16));
  _Float16*  c2w2t = (_Float16*) carve(128 * 64 * sizeof(_Float16));
  _Float16*  c3w1t = (_Float16*) carve(256 * 256 * sizeof(_Float16));
  _Float16*  c3w2t = (_Float16*) carve(512 * 256 * sizeof(_Float16));
  _Float16*  sw1t  = (_Float16*) carve(256 * 672 * sizeof(_Float16));
  _Float16*  sw2t  = (_Float16*) carve(32 * 256 * sizeof(_Float16));
  _Float16*  ew1t  = (_Float16*) carve(32 * 96 * sizeof(_Float16));
  _Float16*  Ebuf  = (_Float16*) carve((size_t)M * 256 * sizeof(_Float16));
  _Float16*  Hbuf  = (_Float16*) carve((size_t)M * 256 * sizeof(_Float16));
  _Float16*  sfh   = Ebuf;   // (BN,256) reuse
  _Float16*  sf2   = Hbuf;   // (BN,32)  reuse

  float* sig_out   = (float*)d_out;
  float* logit_out = (float*)d_out + (size_t)B * P;

  const dim3 knnGrid((N + 127) / 128, B);
  auto wgrid = [](int kp, int np) { return dim3((np * kp + 255) / 256); };

  // ---- one-time weight pad/transpose/convert to f16 ----
  convert_wT_kernel<<<wgrid(32, 32),   256, 0, stream>>>(c1_w1, 6, 16,   c1w1t, 32, 32);
  convert_wT_kernel<<<wgrid(32, 32),   256, 0, stream>>>(c1_w2, 16, 32,  c1w2t, 32, 32);
  convert_wT_kernel<<<wgrid(64, 64),   256, 0, stream>>>(c2_w1, 64, 64,  c2w1t, 64, 64);
  convert_wT_kernel<<<wgrid(64, 128),  256, 0, stream>>>(c2_w2, 64, 128, c2w2t, 64, 128);
  convert_wT_kernel<<<wgrid(256, 256), 256, 0, stream>>>(c3_w1, 256, 256, c3w1t, 256, 256);
  convert_wT_kernel<<<wgrid(256, 512), 256, 0, stream>>>(c3_w2, 256, 512, c3w2t, 256, 512);
  convert_wT_kernel<<<wgrid(672, 256), 256, 0, stream>>>(s_w1, 672, 256, sw1t, 672, 256);
  convert_wT_kernel<<<wgrid(256, 32),  256, 0, stream>>>(s_w2, 256, 32,  sw2t, 256, 32);
  convert_wT_kernel<<<wgrid(96, 32),   256, 0, stream>>>(e_w1, 96, 32,   ew1t, 96, 32);
  pad_f32_kernel<<<1, 64, 0, stream>>>(c1_b1, 1, 16, c1b1p, 1, 32);

  // ---- EdgeConv 1: pos(3) -> 16 -> 32  (K padded 6->32, N padded 16->32) ----
  knn_kernel<3><<<knnGrid, 128, 0, stream>>>(pos, 3, idx, N);
  build_edge_kernel<<<M, 64, 0, stream>>>(pos, 3, idx, Ebuf, N, 3, 32);
  gemm_wmma_kernel<true, false, _Float16, _Float16>
      <<<dim3(M / 32, 1), 32, 0, stream>>>(Ebuf, 32, c1w1t, 32, c1b1p,
                                           Hbuf, 32, M, 32);
  gemm_wmma_kernel<false, true, _Float16, float>
      <<<dim3(M / 32, 1), 32, 0, stream>>>(Hbuf, 32, c1w2t, 32, c1_b2,
                                           xcat + 0, 672, M, 32);

  // ---- EdgeConv 2: x1(32) -> 64 -> 128 ----
  knn_kernel<32><<<knnGrid, 128, 0, stream>>>(xcat + 0, 672, idx, N);
  build_edge_kernel<<<M, 64, 0, stream>>>(xcat + 0, 672, idx, Ebuf, N, 32, 64);
  gemm_wmma_kernel<true, false, _Float16, _Float16>
      <<<dim3(M / 32, 2), 32, 0, stream>>>(Ebuf, 64, c2w1t, 64, c2_b1,
                                           Hbuf, 64, M, 64);
  gemm_wmma_kernel<false, true, _Float16, float>
      <<<dim3(M / 32, 4), 32, 0, stream>>>(Hbuf, 64, c2w2t, 64, c2_b2,
                                           xcat + 32, 672, M, 64);

  // ---- EdgeConv 3: x2(128) -> 256 -> 512 ----
  knn_kernel<128><<<knnGrid, 128, 0, stream>>>(xcat + 32, 672, idx, N);
  build_edge_kernel<<<M, 64, 0, stream>>>(xcat + 32, 672, idx, Ebuf, N, 128, 256);
  gemm_wmma_kernel<true, false, _Float16, _Float16>
      <<<dim3(M / 32, 8), 32, 0, stream>>>(Ebuf, 256, c3w1t, 256, c3_b1,
                                           Hbuf, 256, M, 256);
  gemm_wmma_kernel<false, true, _Float16, float>
      <<<dim3(M / 32, 16), 32, 0, stream>>>(Hbuf, 256, c3w2t, 256, c3_b2,
                                            xcat + 160, 672, M, 256);

  // ---- Scoring MLP: 672 -> 256 (relu) -> 32 ----
  gemm_wmma_kernel<true, false, float, _Float16>
      <<<dim3(BN / 32, 8), 32, 0, stream>>>(xcat, 672, sw1t, 672, s_b1,
                                            sfh, 256, BN, 672);
  gemm_wmma_kernel<false, false, _Float16, _Float16>
      <<<dim3(BN / 32, 1), 32, 0, stream>>>(sfh, 256, sw2t, 256, s_b2,
                                            sf2, 32, BN, 256);

  // ---- Global max over points ----
  colmax_kernel<<<1, 128, 0, stream>>>(sf2, gbuf, N);

  // ---- Pairwise edge classifier ----
  edge_head_kernel<<<dim3((P + 15) / 16, B), 32, 0, stream>>>(
      sf2, gbuf, ew1t, e_b1, e_w2, e_b2, sig_out, logit_out, N, P);
}
